// MultiAttention_30150670418335
// MI455X (gfx1250) — matrix operand order repaired and checked by reading, not compiled, not run on hardware
//
#include <hip/hip_runtime.h>

typedef __attribute__((ext_vector_type(16))) _Float16 v16h;
typedef __attribute__((ext_vector_type(8)))  _Float16 v8h;
typedef __attribute__((ext_vector_type(8)))  float    v8f;

#define LRELU_ALPHA 0.2f
#define NEG_INF (-9000000000000000.0f)

union AFrag { v16h v; v8h h2[2]; _Float16 e[16]; };
union CFrag { v8f v; float f[8]; };
union F4    { float4 v; float f[4]; };

__device__ __forceinline__ float lrelu(float x){ return x > 0.f ? x : LRELU_ALPHA * x; }
__device__ __forceinline__ float eluf (float x){ return x > 0.f ? x : (__expf(x) - 1.f); }

// swap data between lane L and L+16 (SWAPX16): offset = and 0x1f | xor 0x10<<10
__device__ __forceinline__ float swapx16(float x){
  return __builtin_bit_cast(float, __builtin_amdgcn_ds_swizzle(__builtin_bit_cast(int, x), 0x401f));
}
__device__ __forceinline__ float bperm_lane(int srcLane, float x){
  return __builtin_bit_cast(float, __builtin_amdgcn_ds_bpermute(srcLane << 2, __builtin_bit_cast(int, x)));
}

// ---------------------------------------------------------------- mask packing
__global__ void __launch_bounds__(256) k_pack_mask(const int* __restrict__ adj,
                                                   unsigned* __restrict__ mb){
  int idx = blockIdx.x * 256 + threadIdx.x;
  bool p = adj[idx] > 0;
#if __has_builtin(__builtin_amdgcn_ballot_w32)
  unsigned w = __builtin_amdgcn_ballot_w32(p);
#else
  unsigned w = (unsigned)__ballot(p);
#endif
  if ((threadIdx.x & 31) == 0) mb[idx >> 5] = w;
}

// ------------------------------------------- weights: f32 [h][K][Fo] -> f16 [h][Fo][K]
__global__ void __launch_bounds__(256) k_cvt_wt(const float* __restrict__ W,
                                                _Float16* __restrict__ WT,
                                                int K, int Fo, int total){
  int idx = blockIdx.x * 256 + threadIdx.x;
  if (idx >= total) return;
  int k = idx % K; int t = idx / K; int f = t % Fo; int h = t / Fo;
  WT[idx] = (_Float16)W[((size_t)h * K + k) * Fo + f];
}

// ---------------------------------------------------------------- WMMA GEMM
// One wave owns a full 16-row x (NT*16)-col strip: A fragment reused NT times.
// C[h][M][Fo] = A[M][K] @ WT[h][Fo][K]^T, Fo = NT*16, K templated for full unroll.
template<bool AF16, int KVAL, int NT>
__global__ void __launch_bounds__(256) k_gemm(const void* __restrict__ Aptr,
                                              const _Float16* __restrict__ WT,
                                              float* __restrict__ C,
                                              int M, int tilesPerHead){
  constexpr int Fo = NT * 16;
  int wid  = blockIdx.x * 8 + (threadIdx.x >> 5);
  int lane = threadIdx.x & 31;
  int h    = wid / tilesPerHead;
  int tm   = wid % tilesPerHead;
  int row  = tm * 16 + (lane & 15);
  int c0    = (lane < 16) ? 0 : 8;    // A-fragment K sub-offset
  int koffB = (lane < 16) ? 0 : 8;    // B-fragment K sub-offset
  const _Float16* wbase = WT + (size_t)h * Fo * KVAL + (size_t)(lane & 15) * KVAL;

  CFrag acc[NT];
  #pragma unroll
  for (int nt = 0; nt < NT; ++nt)
    #pragma unroll
    for (int i = 0; i < 8; ++i) acc[nt].f[i] = 0.f;

  #pragma unroll
  for (int k0 = 0; k0 < KVAL; k0 += 32){
    AFrag a;
    if (AF16){
      const _Float16* Ah = (const _Float16*)Aptr + (size_t)row * KVAL + k0 + c0;
      a.h2[0] = *(const v8h*)Ah;
      a.h2[1] = *(const v8h*)(Ah + 16);
    } else {
      const float* Af = (const float*)Aptr + (size_t)row * KVAL + k0 + c0;
      F4 x0, x1, x2, x3;
      x0.v = *(const float4*)Af;        x1.v = *(const float4*)(Af + 4);
      x2.v = *(const float4*)(Af + 16); x3.v = *(const float4*)(Af + 20);
      #pragma unroll
      for (int i = 0; i < 4; ++i){
        a.e[i]      = (_Float16)x0.f[i];
        a.e[4 + i]  = (_Float16)x1.f[i];
        a.e[8 + i]  = (_Float16)x2.f[i];
        a.e[12 + i] = (_Float16)x3.f[i];
      }
    }
    #pragma unroll
    for (int nt = 0; nt < NT; ++nt){
      const _Float16* wp = wbase + (size_t)nt * 16 * KVAL + k0 + koffB;
      AFrag b;
      b.h2[0] = *(const v8h*)wp;
      b.h2[1] = *(const v8h*)(wp + 16);
      acc[nt].v = __builtin_amdgcn_wmma_f32_16x16x32_f16(false, a.v, false, b.v,
                                                         (short)0, acc[nt].v, false, false);
    }
  }
  int rbase = tm * 16 + ((lane >> 4) << 3);
  float* Crow = C + (size_t)h * M * Fo;
  #pragma unroll
  for (int r = 0; r < 8; ++r)
    #pragma unroll
    for (int nt = 0; nt < NT; ++nt)
      Crow[(size_t)(rbase + r) * Fo + nt * 16 + (lane & 15)] = acc[nt].f[r];
}

// --------------------------------------------- e_src / e_dst : Wh row . a halves
__global__ void __launch_bounds__(256) k_esrc(const float* __restrict__ Wh,
                                              const float* __restrict__ a,
                                              float* __restrict__ es, float* __restrict__ ed,
                                              int Fo, int M){
  int gid = blockIdx.x * 256 + threadIdx.x;
  int h = gid / M;
  const float* row = Wh + (size_t)gid * Fo;
  const float* ah  = a + (size_t)h * 2 * Fo;
  float s = 0.f, d = 0.f;
  for (int f = 0; f < Fo; f += 4){
    F4 w; w.v = *(const float4*)(row + f);
    #pragma unroll
    for (int i = 0; i < 4; ++i){ s += w.f[i] * ah[f + i]; d += w.f[i] * ah[Fo + f + i]; }
  }
  es[gid] = s; ed[gid] = d;
}

// ------------------------------------------------- per (h,b) max_j e_dst
__global__ void __launch_bounds__(256) k_dmax(const float* __restrict__ ed,
                                              float* __restrict__ dmax){
  __shared__ float red[256];
  const float* p = ed + (size_t)blockIdx.x * 2048;
  float m = -3.4e38f;
  for (int j = threadIdx.x; j < 2048; j += 256) m = fmaxf(m, p[j]);
  red[threadIdx.x] = m; __syncthreads();
  for (int s = 128; s > 0; s >>= 1){
    if (threadIdx.x < s) red[threadIdx.x] = fmaxf(red[threadIdx.x], red[threadIdx.x + s]);
    __syncthreads();
  }
  if (threadIdx.x == 0) dmax[blockIdx.x] = red[0];
}

// -------------------------------- Wh f32 [hb][2048][Fo] -> WhT f16 [hb][Fo][2048]
__global__ void __launch_bounds__(256) k_transpose(const float* __restrict__ Wh,
                                                   _Float16* __restrict__ WhT, int Fo){
  __shared__ _Float16 t[64 * 65];
  int tile = blockIdx.x & 31;
  int hb   = blockIdx.x >> 5;
  const float* src = Wh + ((size_t)hb * 2048 + tile * 64) * Fo;
  for (int idx = threadIdx.x; idx < 64 * Fo; idx += 256){
    int r = idx / Fo, f = idx % Fo;
    t[r * 65 + f] = (_Float16)src[(size_t)r * Fo + f];
  }
  __syncthreads();
  _Float16* dst = WhT + (size_t)hb * Fo * 2048 + tile * 64;
  for (int idx = threadIdx.x; idx < 64 * Fo; idx += 256){
    int f = idx >> 6, r = idx & 63;
    dst[(size_t)f * 2048 + r] = t[r * 65 + f];
  }
}

// ---------------------------------------------------------------- GAT attention
// One wave = one 16-row tile of one (head,batch). Fo = 64.
// Logits recomputed from e_src/e_dst (rank-1); single pass softmax with
// analytic max bound m_i = lrelu(s_i + max_j d_j); P (f16) @ WhT via WMMA.
__global__ void __launch_bounds__(256) k_attn(const _Float16* __restrict__ whT,
                                              const float* __restrict__ es,
                                              const float* __restrict__ ed,
                                              const float* __restrict__ dmax,
                                              const unsigned* __restrict__ mb,
                                              _Float16* __restrict__ out16, int stride16,
                                              int colPerHead,
                                              float* __restrict__ out32){
  int wid  = blockIdx.x * 8 + (threadIdx.x >> 5);
  int lane = threadIdx.x & 31;
  int hb   = wid >> 7;           // (h*B + b)
  int tile = wid & 127;
  int b = hb & 3;
  int h = hb >> 2;
  int i0  = tile << 4;
  int row = i0 + (lane & 15);

  float s = es[(size_t)hb * 2048 + row];
  float m = lrelu(s + dmax[hb]);               // >= true row max (lrelu monotone)
  int c0   = (lane < 16) ? 0 : 8;
  int koff = (lane < 16) ? 0 : 8;
  const float*    edb  = ed + (size_t)hb * 2048;
  const unsigned* mrow = mb + (size_t)row * 64;
  const _Float16* wbase = whT + (size_t)hb * 64 * 2048;

  CFrag acc[4];
  #pragma unroll
  for (int nt = 0; nt < 4; ++nt)
    #pragma unroll
    for (int i = 0; i < 8; ++i) acc[nt].f[i] = 0.f;

  float l = 0.f;
  for (int j0 = 0; j0 < 2048; j0 += 32){
    F4 d0, d1, d2, d3;
    d0.v = *(const float4*)(edb + j0 + c0);
    d1.v = *(const float4*)(edb + j0 + c0 + 4);
    d2.v = *(const float4*)(edb + j0 + c0 + 16);
    d3.v = *(const float4*)(edb + j0 + c0 + 20);
    unsigned w = mrow[j0 >> 5];

    AFrag a;
    #pragma unroll
    for (int k = 0; k < 8; ++k){
      float dlo = (k < 4) ? d0.f[k] : d1.f[k - 4];
      float dhi = (k < 4) ? d2.f[k] : d3.f[k - 4];
      float e0 = lrelu(s + dlo);
      e0 = ((w >> (c0 + k)) & 1u) ? e0 : NEG_INF;
      float p0 = __expf(e0 - m);
      float e1 = lrelu(s + dhi);
      e1 = ((w >> (c0 + 16 + k)) & 1u) ? e1 : NEG_INF;
      float p1 = __expf(e1 - m);
      l += p0 + p1;
      a.e[k]     = (_Float16)p0;
      a.e[8 + k] = (_Float16)p1;
    }
    #pragma unroll
    for (int nt = 0; nt < 4; ++nt){
      const _Float16* wp = wbase + (size_t)(nt * 16 + (lane & 15)) * 2048 + j0 + koff;
      AFrag bf;
      bf.h2[0] = *(const v8h*)wp;
      bf.h2[1] = *(const v8h*)(wp + 16);
      __builtin_prefetch((const void*)(wp + 32), 0, 3);   // next K chunk, WGP scope
      acc[nt].v = __builtin_amdgcn_wmma_f32_16x16x32_f16(false, a.v, false, bf.v,
                                                         (short)0, acc[nt].v, false, false);
    }
  }

  // combine row sum across half-wave pair; broadcast 1/l into C-fragment rows
  l += swapx16(l);
  float invl = __builtin_amdgcn_rcpf(fmaxf(l, 1e-30f));   // v_rcp_f32, no IEEE divide
  int hi8  = (lane >> 4) << 3;
  int colb = colPerHead * h + (lane & 15);
  #pragma unroll
  for (int r = 0; r < 8; ++r){
    float sr = bperm_lane(r + hi8, invl);    // 1/l of row (r + hi8)
    int gi = b * 2048 + i0 + r + hi8;
    #pragma unroll
    for (int nt = 0; nt < 4; ++nt){
      float o = eluf(acc[nt].f[r] * sr);
      out16[(size_t)gi * stride16 + colb + nt * 16] = (_Float16)o;
      if (out32) out32[(size_t)gi * 64 + (lane & 15) + nt * 16] = o;
    }
  }
}

// -------------------------------------- layer 3: only attention row 0 is needed
__global__ void __launch_bounds__(256) k_final(const float* __restrict__ Wh3,
                                               const float* __restrict__ es,
                                               const float* __restrict__ ed,
                                               const unsigned* __restrict__ mb,
                                               float* __restrict__ pre){
  int b = blockIdx.x, tid = threadIdx.x;
  __shared__ float red[256];
  __shared__ float accS[17];
  float s0 = es[(size_t)b * 2048];            // i = 0
  float m = -3.4e38f;
  for (int j = tid; j < 2048; j += 256){
    float e = lrelu(s0 + ed[(size_t)b * 2048 + j]);
    e = ((mb[j >> 5] >> (j & 31)) & 1u) ? e : NEG_INF;
    m = fmaxf(m, e);
  }
  red[tid] = m; __syncthreads();
  for (int s = 128; s > 0; s >>= 1){
    if (tid < s) red[tid] = fmaxf(red[tid], red[tid + s]);
    __syncthreads();
  }
  m = red[0];
  if (tid < 17) accS[tid] = 0.f;
  __syncthreads();

  float l = 0.f;
  float acc[16];
  #pragma unroll
  for (int f = 0; f < 16; ++f) acc[f] = 0.f;
  for (int j = tid; j < 2048; j += 256){
    float e = lrelu(s0 + ed[(size_t)b * 2048 + j]);
    bool bit = ((mb[j >> 5] >> (j & 31)) & 1u);
    float p = bit ? __expf(e - m) : 0.f;
    l += p;
    const float* wr = Wh3 + (size_t)(b * 2048 + j) * 16;
    #pragma unroll
    for (int f = 0; f < 16; f += 4){
      F4 wv; wv.v = *(const float4*)(wr + f);
      #pragma unroll
      for (int i = 0; i < 4; ++i) acc[f + i] += p * wv.f[i];
    }
  }
  atomicAdd(&accS[16], l);
  #pragma unroll
  for (int f = 0; f < 16; ++f) atomicAdd(&accS[f], acc[f]);
  __syncthreads();
  if (tid < 16) pre[b * 16 + tid] = eluf(accS[tid] / fmaxf(accS[16], 1e-30f));
}

// ================================================================ launch
extern "C" void kernel_launch(void* const* d_in, const int* in_sizes, int n_in,
                              void* d_out, int out_size, void* d_ws, size_t ws_size,
                              hipStream_t stream){
  (void)in_sizes; (void)n_in; (void)out_size; (void)ws_size;
  const float* slices = (const float*)d_in[0];   // [4,2048,128]
  const int*   adj    = (const int*)  d_in[1];   // [2048,2048]
  const float* Ws     = (const float*)d_in[2];   // [8,128,64]
  const float* As     = (const float*)d_in[3];   // [8,128,1]
  const float* W1     = (const float*)d_in[4];   // [512,64]
  const float* a1     = (const float*)d_in[5];   // [128,1]
  const float* W2     = (const float*)d_in[6];   // [64,16]
  const float* a2     = (const float*)d_in[7];   // [32,1]
  float* image_feature = (float*)d_out;                    // [4,2048,64]
  float* pre           = (float*)d_out + 4 * 2048 * 64;    // [4,16]

  char* ws = (char*)d_ws;
  size_t off = 0;
  auto alloc = [&](size_t bytes) -> char* {
    char* p = ws + off;
    off = (off + bytes + 255) & ~(size_t)255;
    return p;
  };
  unsigned*  maskbits = (unsigned*) alloc((size_t)2048 * 64 * 4);
  _Float16*  WT1      = (_Float16*) alloc((size_t)8 * 64 * 128 * 2);
  _Float16*  WT2      = (_Float16*) alloc((size_t)64 * 512 * 2);
  _Float16*  WT3      = (_Float16*) alloc((size_t)16 * 64 * 2);
  float*     Wh1      = (float*)    alloc((size_t)8 * 8192 * 64 * 4);
  float*     es1      = (float*)    alloc((size_t)8 * 8192 * 4);
  float*     ed1      = (float*)    alloc((size_t)8 * 8192 * 4);
  float*     dm1      = (float*)    alloc(32 * 4);
  _Float16*  WhT1     = (_Float16*) alloc((size_t)32 * 64 * 2048 * 2);
  _Float16*  x16      = (_Float16*) alloc((size_t)8192 * 512 * 2);
  float*     Wh2      = (float*)    alloc((size_t)8192 * 64 * 4);
  float*     es2      = (float*)    alloc((size_t)8192 * 4);
  float*     ed2      = (float*)    alloc((size_t)8192 * 4);
  float*     dm2      = (float*)    alloc(4 * 4);
  _Float16*  WhT2     = (_Float16*) alloc((size_t)4 * 64 * 2048 * 2);
  _Float16*  img16    = (_Float16*) alloc((size_t)8192 * 64 * 2);
  float*     Wh3      = (float*)    alloc((size_t)8192 * 16 * 4);
  float*     es3      = (float*)    alloc((size_t)8192 * 4);
  float*     ed3      = (float*)    alloc((size_t)8192 * 4);

  // setup
  k_pack_mask<<<16384, 256, 0, stream>>>(adj, maskbits);
  k_cvt_wt<<<256, 256, 0, stream>>>(Ws, WT1, 128, 64, 8 * 128 * 64);
  k_cvt_wt<<<128, 256, 0, stream>>>(W1, WT2, 512, 64, 512 * 64);
  k_cvt_wt<<<4,   256, 0, stream>>>(W2, WT3, 64, 16, 64 * 16);

  // ---- layer 1 (8 heads): 8 x [8192x64x128] GEMM, waves = 8*512 -> 512 blocks
  k_gemm<false, 128, 4><<<512, 256, 0, stream>>>(slices, WT1, Wh1, 8192, 512);
  k_esrc<<<256, 256, 0, stream>>>(Wh1, As, es1, ed1, 64, 8192);
  k_dmax<<<32, 256, 0, stream>>>(ed1, dm1);
  k_transpose<<<1024, 256, 0, stream>>>(Wh1, WhT1, 64);
  k_attn<<<512, 256, 0, stream>>>(WhT1, es1, ed1, dm1, maskbits, x16, 512, 64, nullptr);

  // ---- layer 2: [8192x64x512] GEMM, waves = 512 -> 64 blocks
  k_gemm<true, 512, 4><<<64, 256, 0, stream>>>(x16, WT2, Wh2, 8192, 512);
  k_esrc<<<32, 256, 0, stream>>>(Wh2, a1, es2, ed2, 64, 8192);
  k_dmax<<<4, 256, 0, stream>>>(ed2, dm2);
  k_transpose<<<128, 256, 0, stream>>>(Wh2, WhT2, 64);
  k_attn<<<64, 256, 0, stream>>>(WhT2, es2, ed2, dm2, maskbits, img16, 64, 0, image_feature);

  // ---- layer 3: [8192x16x64] GEMM, waves = 512 -> 64 blocks
  k_gemm<true, 64, 1><<<64, 256, 0, stream>>>(img16, WT3, Wh3, 8192, 512);
  k_esrc<<<32, 256, 0, stream>>>(Wh3, a2, es3, ed3, 16, 8192);
  k_final<<<4, 256, 0, stream>>>(Wh3, es3, ed3, maskbits, pre);
}
